// LSTM_81681688035993
// MI455X (gfx1250) — compile-verified, matrix operand order checked by reading, not run on hardware
//
#include <hip/hip_runtime.h>
#include <cstddef>

// ---------------------------------------------------------------------------
// 2-layer LSTM (B=256, T=2048, D=H=64) + final FC(64->128)+ReLU on last h.
// Persistent-recurrence kernel for gfx1250 (CDNA5, wave32, WMMA).
//
// grid = 16 workgroups, each owns a 16-row batch tile.
// block = 256 threads = 8 wave32 waves; wave w owns gate columns [32w, 32w+32).
// Weights (f16) live in VGPRs; bias folded into WMMA accumulators; h/c state
// on-chip; x software-pipelined + prefetched; 4 barriers per timestep.
// ---------------------------------------------------------------------------

typedef __attribute__((ext_vector_type(16))) _Float16 v16h;
typedef __attribute__((ext_vector_type(8)))  _Float16 h8;
typedef __attribute__((ext_vector_type(8)))  float    v8f;

#define LSTM_T 2048
#define LSTM_D 64
#define LSTM_H 64

#ifndef __has_builtin
#define __has_builtin(x) 0
#endif

#if __has_builtin(__builtin_amdgcn_tanhf)
// gfx1250 hardware v_tanh_f32 (single TRANS op)
__device__ __forceinline__ float tanh_f(float x) { return __builtin_amdgcn_tanhf(x); }
__device__ __forceinline__ float sigm(float x)   { return fmaf(0.5f, __builtin_amdgcn_tanhf(0.5f * x), 0.5f); }
#else
// branch-free fallback on raw v_exp_f32 / v_rcp_f32
__device__ __forceinline__ float sigm(float x) {
    return __builtin_amdgcn_rcpf(1.0f + __builtin_amdgcn_exp2f(-1.442695041f * x));
}
__device__ __forceinline__ float tanh_f(float x) {
    // tanh(x) = 1 - 2/(exp(2x)+1); exact limits at +-inf via rcp(inf)=0
    return 1.0f - 2.0f * __builtin_amdgcn_rcpf(1.0f + __builtin_amdgcn_exp2f(2.885390082f * x));
}
#endif

#define WMMA_F16(a, b, c) \
    __builtin_amdgcn_wmma_f32_16x16x32_f16(false, (a), false, (b), (short)0, (c), false, false)

__global__ __launch_bounds__(256, 1)
void lstm2_persistent_kernel(const float* __restrict__ x,
                             const float* __restrict__ Wih0, const float* __restrict__ Whh0,
                             const float* __restrict__ bih0, const float* __restrict__ bhh0,
                             const float* __restrict__ Wih1, const float* __restrict__ Whh1,
                             const float* __restrict__ bih1, const float* __restrict__ bhh1,
                             const float* __restrict__ fcW,  const float* __restrict__ fcb,
                             float* __restrict__ out)
{
    __shared__ __align__(16) _Float16 xbuf [16 * 64];   // layer-0 input tile (f16)
    __shared__ __align__(16) _Float16 h0buf[16 * 64];   // layer-0 hidden (== layer-1 input)
    __shared__ __align__(16) _Float16 h1buf[16 * 64];   // layer-1 hidden
    __shared__ __align__(16) float    gates[16 * 256];  // per-layer post-bias gate pre-activations
    __shared__ __align__(16) float    hlast[16 * 64];   // fp32 h1 at t = T-1

    const int tid  = threadIdx.x;
    const int lane = tid & 31;
    const int wave = tid >> 5;
    const int B0   = blockIdx.x * 16;        // batch tile base

    // ---- one-time setup ---------------------------------------------------
    // zero-init hidden buffers (1024 halfs each, 4 per thread)
    {
        int e = tid * 4;
        #pragma unroll
        for (int k = 0; k < 4; ++k) {
            h0buf[e + k] = (_Float16)0.0f;
            h1buf[e + k] = (_Float16)0.0f;
        }
    }

    // Per-lane bias values for this lane's gate columns (folded into WMMA C).
    // acc element (j, lane) -> gate column  wave*32 + (lane&15)  (tile0), +16 (tile1)
    const int gcol = wave * 32 + (lane & 15);
    const float b0a = bih0[gcol]      + bhh0[gcol];
    const float b0b = bih0[gcol + 16] + bhh0[gcol + 16];
    const float b1a = bih1[gcol]      + bhh1[gcol];
    const float b1b = bih1[gcol + 16] + bhh1[gcol + 16];

    // Weight fragments in VGPRs, ISA 16-bit B layout (== A layout with N as row):
    // lane l: row = nbase + (l%16); elems 0..7 -> K = kbase + koff + j,
    //         elems 8..15 -> K = kbase + 16 + koff + j, koff = (l/16)*8.
    // Wcat row g = [Wih[g, 0:64] | Whh[g, 0:64]], kt 0..1 from Wih, kt 2..3 from Whh.
    v16h wf0[2][4];
    v16h wf1[2][4];
    {
        const int koff = (lane >> 4) << 3;
        #pragma unroll
        for (int n = 0; n < 2; ++n) {
            const int row = wave * 32 + n * 16 + (lane & 15);
            #pragma unroll
            for (int kt = 0; kt < 4; ++kt) {
                const float* src = (kt < 2) ? (Wih0 + row * 64 + kt * 32)
                                            : (Whh0 + row * 64 + (kt - 2) * 32);
                v16h f;
                #pragma unroll
                for (int j = 0; j < 8; ++j) {
                    f[j]     = (_Float16)src[koff + j];
                    f[8 + j] = (_Float16)src[16 + koff + j];
                }
                wf0[n][kt] = f;
                src = (kt < 2) ? (Wih1 + row * 64 + kt * 32)
                               : (Whh1 + row * 64 + (kt - 2) * 32);
                #pragma unroll
                for (int j = 0; j < 8; ++j) {
                    f[j]     = (_Float16)src[koff + j];
                    f[8 + j] = (_Float16)src[16 + koff + j];
                }
                wf1[n][kt] = f;
            }
        }
    }

    // per-thread cell state: thread owns (b = tid/16, hid = (tid%16)*4 .. +3)
    float c0[4] = {0.f, 0.f, 0.f, 0.f};
    float c1[4] = {0.f, 0.f, 0.f, 0.f};

    // A-fragment load from an LDS 16x64 f16 row-major tile (ISA 16-bit A layout)
    auto loadA = [&](const _Float16* buf, int fbase) -> v16h {
        const int r    = lane & 15;
        const int koff = (lane >> 4) << 3;
        h8 lo = *(const h8*)(buf + r * 64 + fbase + koff);
        h8 hi = *(const h8*)(buf + r * 64 + fbase + 16 + koff);
        v16h a;
        #pragma unroll
        for (int j = 0; j < 8; ++j) { a[j] = lo[j]; a[8 + j] = hi[j]; }
        return a;
    };

    // gates(16x256) = [xt|h](16x128) x Wcat^T + bias, via 8 WMMAs/wave.
    // All 4 A fragments issued up front so ds_load_b128s pipeline under WMMA issue.
    auto cell_gemm = [&](const _Float16* xb, const _Float16* hb,
                         const v16h (&wf)[2][4], float ba, float bb) {
        v16h av0 = loadA(xb, 0);
        v16h av1 = loadA(xb, 32);
        v16h av2 = loadA(hb, 0);
        v16h av3 = loadA(hb, 32);
        v8f acc0 = {ba, ba, ba, ba, ba, ba, ba, ba};
        v8f acc1 = {bb, bb, bb, bb, bb, bb, bb, bb};
        acc0 = WMMA_F16(av0, wf[0][0], acc0);
        acc1 = WMMA_F16(av0, wf[1][0], acc1);
        acc0 = WMMA_F16(av1, wf[0][1], acc0);
        acc1 = WMMA_F16(av1, wf[1][1], acc1);
        acc0 = WMMA_F16(av2, wf[0][2], acc0);
        acc1 = WMMA_F16(av2, wf[1][2], acc1);
        acc0 = WMMA_F16(av3, wf[0][3], acc0);
        acc1 = WMMA_F16(av3, wf[1][3], acc1);
        // scatter C tiles to LDS gates: elem j of lane l -> (M = j + 8*(l>>4), N = l&15)
        const int r0   = (lane >> 4) << 3;
        const int ncol = wave * 32 + (lane & 15);
        #pragma unroll
        for (int j = 0; j < 8; ++j) {
            gates[(r0 + j) * 256 + ncol]      = acc0[j];
            gates[(r0 + j) * 256 + ncol + 16] = acc1[j];
        }
    };

    // elementwise LSTM cell update for this thread's 4 (b, hid) elements
    auto cell_update = [&](float* c, _Float16* hb, bool storeLast) {
        const int b    = tid >> 4;
        const int hbse = (tid & 15) << 2;
        const float* grow = &gates[b * 256];
        float4 iv = *(const float4*)(grow + hbse);
        float4 fv = *(const float4*)(grow + 64 + hbse);
        float4 gv = *(const float4*)(grow + 128 + hbse);
        float4 ov = *(const float4*)(grow + 192 + hbse);
        float ia[4] = {iv.x, iv.y, iv.z, iv.w};
        float fa[4] = {fv.x, fv.y, fv.z, fv.w};
        float ga[4] = {gv.x, gv.y, gv.z, gv.w};
        float oa[4] = {ov.x, ov.y, ov.z, ov.w};
        #pragma unroll
        for (int k = 0; k < 4; ++k) {
            const float ii = sigm(ia[k]);
            const float ff = sigm(fa[k]);
            const float gg = tanh_f(ga[k]);
            const float oo = sigm(oa[k]);
            c[k] = ff * c[k] + ii * gg;
            const float h = oo * tanh_f(c[k]);
            hb[b * 64 + hbse + k] = (_Float16)h;
            if (storeLast) hlast[b * 64 + hbse + k] = h;
        }
    };

    // ---- time loop (software-pipelined x streaming) -----------------------
    const int xb_row = tid >> 4;            // 0..15  (batch row within tile)
    const int xb_col = (tid & 15) << 2;     // 0..60  (4 features per thread)
    const float* xrow = x + (((size_t)(B0 + xb_row)) * (size_t)LSTM_T) * LSTM_D + xb_col;
    _Float16* xd = &xbuf[xb_row * 64 + xb_col];

    // prologue: publish x[0]
    {
        float4 xv = *(const float4*)xrow;
        xd[0] = (_Float16)xv.x; xd[1] = (_Float16)xv.y;
        xd[2] = (_Float16)xv.z; xd[3] = (_Float16)xv.w;
    }
    __syncthreads();

    for (int t = 0; t < LSTM_T; ++t) {
        // issue next timestep's x load now; consumed at the iteration tail
        float4 xn;
        const bool have_next = (t + 1 < LSTM_T);
        if (have_next)
            xn = *(const float4*)(xrow + (size_t)(t + 1) * LSTM_D);
        if (t + 16 < LSTM_T)
            __builtin_prefetch(xrow + (size_t)(t + 16) * LSTM_D, 0, 1);  // global_prefetch_b8

        // layer 0
        cell_gemm(xbuf, h0buf, wf0, b0a, b0b);
        __syncthreads();
        cell_update(c0, h0buf, false);
        __syncthreads();

        // layer 1 (input = new h0)
        cell_gemm(h0buf, h1buf, wf1, b1a, b1b);
        __syncthreads();
        cell_update(c1, h1buf, t == LSTM_T - 1);

        // publish x[t+1] (xbuf reads for step t finished before the 2nd barrier)
        if (have_next) {
            xd[0] = (_Float16)xn.x; xd[1] = (_Float16)xn.y;
            xd[2] = (_Float16)xn.z; xd[3] = (_Float16)xn.w;
        }
        __syncthreads();
    }

    // ---- final FC: out = relu(hlast @ fcW^T + fcb), fp32 ------------------
    {
        const int b  = tid >> 4;            // 0..15
        const int jb = (tid & 15) << 3;     // 8 outputs per thread (128 cols)
        const float* hrow = &hlast[b * 64];
        #pragma unroll
        for (int jj = 0; jj < 8; ++jj) {
            const int j = jb + jj;
            const float* wrow = fcW + j * 64;
            float acc = fcb[j];
            #pragma unroll 16
            for (int k = 0; k < 64; ++k) acc = fmaf(hrow[k], wrow[k], acc);
            out[(size_t)(B0 + b) * 128 + j] = acc > 0.f ? acc : 0.f;
        }
    }
}

extern "C" void kernel_launch(void* const* d_in, const int* in_sizes, int n_in,
                              void* d_out, int out_size, void* d_ws, size_t ws_size,
                              hipStream_t stream) {
    (void)in_sizes; (void)n_in; (void)out_size; (void)d_ws; (void)ws_size;
    const float* x    = (const float*)d_in[0];
    const float* Wih0 = (const float*)d_in[1];
    const float* Whh0 = (const float*)d_in[2];
    const float* bih0 = (const float*)d_in[3];
    const float* bhh0 = (const float*)d_in[4];
    const float* Wih1 = (const float*)d_in[5];
    const float* Whh1 = (const float*)d_in[6];
    const float* bih1 = (const float*)d_in[7];
    const float* bhh1 = (const float*)d_in[8];
    const float* fcW  = (const float*)d_in[9];
    const float* fcb  = (const float*)d_in[10];
    float* out = (float*)d_out;

    hipLaunchKernelGGL(lstm2_persistent_kernel,
                       dim3(16), dim3(256), 0, stream,
                       x, Wih0, Whh0, bih0, bhh0, Wih1, Whh1, bih1, bhh1,
                       fcW, fcb, out);
}
